// UpFormerBlock_linear_attn_29033978921044
// MI455X (gfx1250) — matrix operand order repaired and checked by reading, not compiled, not run on hardware
//
#include <hip/hip_runtime.h>
#include <hip/hip_bf16.h>

typedef __attribute__((ext_vector_type(16))) _Float16 v16h;
typedef __attribute__((ext_vector_type(8)))  _Float16 v8h;
typedef __attribute__((ext_vector_type(8)))  float    v8f;

#define WMMA_F16(a, b, c) \
  __builtin_amdgcn_wmma_f32_16x16x32_f16(false, (a), false, (b), (short)0, (c), false, false)

static constexpr int   NPTS = 120000;
static constexpr int   NT   = NPTS / 16;   // 7500 row tiles
static constexpr float EPS_A  = 1e-4f;
static constexpr float BN_EPS = 1e-3f;
static constexpr float LN_EPS = 1e-3f;

__device__ __forceinline__ v16h pack16(v8h lo, v8h hi) {
  v16h r;
#pragma unroll
  for (int i = 0; i < 8; ++i) { r[i] = lo[i]; r[8 + i] = hi[i]; }
  return r;
}

// A operand (16xK tile, row-major in LDS, strideH halves per row), 32-wide K chunk at kk.
// Lane l: row m = l&15; halves 0..7 = K kk+g*8..+7 ; halves 8..15 = K kk+16+g*8..+7 (g = l>>4).
__device__ __forceinline__ v16h load_opA_lds(const _Float16* tile, int strideH, int kk, int lane) {
  const int m = lane & 15, g = (lane >> 4) & 1;
  const _Float16* p = tile + m * strideH + kk + g * 8;
  return pack16(*(const v8h*)p, *(const v8h*)(p + 16));
}

// B operand from transposed-f16 weights Bt[n][k] (row-major, K halves per row).
// Lane l: col n = n0+(l&15); halves 0..7 = K kk+g*8..+7 ; halves 8..15 = K kk+16+g*8..+7.
__device__ __forceinline__ v16h load_opB_g(const _Float16* Bt, int K, int n0, int kk, int lane) {
  const int n = n0 + (lane & 15), g = (lane >> 4) & 1;
  const _Float16* p = Bt + (size_t)n * K + kk + g * 8;
  return pack16(*(const v8h*)p, *(const v8h*)(p + 16));
}

__device__ __forceinline__ v8f zero8() {
  v8f a;
#pragma unroll
  for (int j = 0; j < 8; ++j) a[j] = 0.0f;
  return a;
}

// ---------- weight transpose + f32->f16 convert: W[b][k][n] -> Bt[b][n][k] ----------
__global__ void conv_wT(const float* __restrict__ W, _Float16* __restrict__ Bt,
                        int K, int Nc, int total) {
  int idx = blockIdx.x * blockDim.x + threadIdx.x;
  if (idx >= total) return;
  int per = K * Nc;
  int b = idx / per, rem = idx - b * per;
  int n = rem / K, k = rem - n * K;
  Bt[(size_t)b * per + (size_t)n * K + k] = (_Float16)W[(size_t)b * per + (size_t)k * Nc + n];
}

// ---------- K1: sp = BN(features @ W_pre + b_pre) ----------
__global__ __launch_bounds__(128) void k1_pre(
    const float* __restrict__ F, const _Float16* __restrict__ WpreT,
    const float* __restrict__ b_pre, const float* __restrict__ g_pre,
    const float* __restrict__ be_pre, const float* __restrict__ m_pre,
    const float* __restrict__ v_pre, float* __restrict__ sp) {
  __shared__ __align__(16) _Float16 Ah[16 * 64];
  const int row0 = blockIdx.x * 16;
  const int t = threadIdx.x;
  {
    int r = t >> 3, c0 = (t & 7) * 8;
    const float* src = F + (size_t)(row0 + r) * 64 + c0;
#pragma unroll
    for (int i = 0; i < 8; ++i) Ah[r * 64 + c0 + i] = (_Float16)src[i];
  }
  __syncthreads();
  const int lane = t & 31, w = t >> 5;
#pragma unroll
  for (int i = 0; i < 2; ++i) {
    const int n0 = (w * 2 + i) * 16;
    v8f acc = zero8();
#pragma unroll
    for (int kk = 0; kk < 64; kk += 32) {
      v16h a = load_opA_lds(Ah, 64, kk, lane);
      v16h b = load_opB_g(WpreT, 64, n0, kk, lane);
      acc = WMMA_F16(a, b, acc);
    }
    const int n = n0 + (lane & 15);
    const float S = g_pre[n] * rsqrtf(v_pre[n] + BN_EPS);
    const float T = (b_pre[n] - m_pre[n]) * S + be_pre[n];
    const int mo = (lane >= 16) ? 8 : 0;
#pragma unroll
    for (int j = 0; j < 8; ++j)
      sp[(size_t)(row0 + j + mo) * 128 + n] = acc[j] * S + T;
  }
}

// ---------- K2: qkv + pe, finalize q/k/v ----------
__global__ __launch_bounds__(128) void k2_qkv(
    const float* __restrict__ sp, const int* __restrict__ indices,
    const float* __restrict__ Wp1, const float* __restrict__ bp1,
    const float* __restrict__ g_pe, const float* __restrict__ be_pe,
    const float* __restrict__ m_pe, const float* __restrict__ v_pe,
    const _Float16* __restrict__ Wp2T, const float* __restrict__ bp2,
    const _Float16* __restrict__ WqkvT, const float* __restrict__ b_qkv,
    const float* __restrict__ scale_p,
    _Float16* __restrict__ Q16, _Float16* __restrict__ K16, _Float16* __restrict__ V16) {
  __shared__ __align__(16) _Float16 Ah[16 * 128];
  __shared__ __align__(16) _Float16 Hh[16 * 128];
  __shared__ float XYZ[16][3];
  __shared__ float QKVf[16 * 384];
  __shared__ float PEf[16 * 128];
  const int row0 = blockIdx.x * 16;
  const int t = threadIdx.x;
  {
    int r = t >> 3, c0 = (t & 7) * 16;
    const float* src = sp + (size_t)(row0 + r) * 128 + c0;
#pragma unroll
    for (int i = 0; i < 16; ++i) Ah[r * 128 + c0 + i] = (_Float16)src[i];
  }
  if (t < 16) {
#pragma unroll
    for (int d = 0; d < 3; ++d)
      XYZ[t][d] = (float)indices[(size_t)(row0 + t) * 4 + 1 + d];
  }
  __syncthreads();
  {  // pe hidden layer (K = 3, VALU)
    const int c = t;
    const float w0 = Wp1[c], w1 = Wp1[128 + c], w2 = Wp1[256 + c];
    const float S = g_pe[c] * rsqrtf(v_pe[c] + BN_EPS);
    const float T = (bp1[c] - m_pe[c]) * S + be_pe[c];
#pragma unroll
    for (int r = 0; r < 16; ++r) {
      float u = XYZ[r][0] * w0 + XYZ[r][1] * w1 + XYZ[r][2] * w2;
      Hh[r * 128 + c] = (_Float16)fmaxf(u * S + T, 0.0f);
    }
  }
  __syncthreads();
  const int lane = t & 31, w = t >> 5;
  const int mo = (lane >= 16) ? 8 : 0;
#pragma unroll
  for (int i = 0; i < 6; ++i) {  // qkv: 24 col tiles
    const int n0 = (w * 6 + i) * 16;
    v8f acc = zero8();
#pragma unroll
    for (int kk = 0; kk < 128; kk += 32) {
      v16h a = load_opA_lds(Ah, 128, kk, lane);
      v16h b = load_opB_g(WqkvT, 128, n0, kk, lane);
      acc = WMMA_F16(a, b, acc);
    }
    const int n = n0 + (lane & 15);
#pragma unroll
    for (int j = 0; j < 8; ++j) QKVf[(j + mo) * 384 + n] = acc[j];
  }
#pragma unroll
  for (int i = 0; i < 2; ++i) {  // pe: 8 col tiles
    const int n0 = (w * 2 + i) * 16;
    v8f acc = zero8();
#pragma unroll
    for (int kk = 0; kk < 128; kk += 32) {
      v16h a = load_opA_lds(Hh, 128, kk, lane);
      v16h b = load_opB_g(Wp2T, 128, n0, kk, lane);
      acc = WMMA_F16(a, b, acc);
    }
    const int n = n0 + (lane & 15);
    const float bb = bp2[n];
#pragma unroll
    for (int j = 0; j < 8; ++j) PEf[(j + mo) * 128 + n] = acc[j] + bb;
  }
  __syncthreads();
  {  // finalize q, k, v
    const int c = t;
    const float bq = b_qkv[c], bk = b_qkv[128 + c], bv = b_qkv[256 + c];
    const float sc = logf(1.0f + expf(scale_p[c]));
    const float inv = 1.0f / sc;
#pragma unroll
    for (int r = 0; r < 16; ++r) {
      float qv = QKVf[r * 384 + c] + bq;
      float kv = QKVf[r * 384 + 128 + c] + bk + PEf[r * 128 + c];
      float vv = QKVf[r * 384 + 256 + c] + bv;
      float q = (fmaxf(qv, 0.0f) + EPS_A) * inv;
      float k = (fmaxf(kv, 0.0f) + EPS_A) * inv;
      size_t o = (size_t)(row0 + r) * 128 + c;
      Q16[o] = (_Float16)q;
      K16[o] = (_Float16)k;
      V16[o] = (_Float16)vv;
    }
  }
}

// ---------- R1: per-block partial kv (WMMA) + ksum ----------
__global__ __launch_bounds__(128) void r1_kv(
    const _Float16* __restrict__ K16, const _Float16* __restrict__ V16,
    float* __restrict__ kvp, float* __restrict__ ksp) {
  __shared__ __align__(16) _Float16 Kh[16 * 128];
  __shared__ __align__(16) _Float16 Vh[16 * 128];
  const int t = threadIdx.x, lane = t & 31, w = t >> 5;
  const int m = lane & 15, g = (lane >> 4) & 1, n = lane & 15;
  v8f acc0 = zero8(), acc1 = zero8();
  float ks = 0.0f;
  for (int tile = blockIdx.x; tile < NT; tile += 64) {
    __syncthreads();
    {
      int r = t >> 3, c0 = (t & 7) * 16;
      size_t o = (size_t)(tile * 16 + r) * 128 + c0;
      *(v8h*)&Kh[r * 128 + c0]     = *(const v8h*)(K16 + o);
      *(v8h*)&Kh[r * 128 + c0 + 8] = *(const v8h*)(K16 + o + 8);
      *(v8h*)&Vh[r * 128 + c0]     = *(const v8h*)(V16 + o);
      *(v8h*)&Vh[r * 128 + c0 + 8] = *(const v8h*)(V16 + o + 8);
    }
    __syncthreads();
#pragma unroll
    for (int r = 0; r < 16; ++r) ks += (float)Kh[r * 128 + t];
#pragma unroll
    for (int hh = 0; hh < 2; ++hh) {
      const int h = w * 2 + hh;
      v16h a;  // A = K_tile^T : A[m][kd] = Kh[kd][h*16+m], K padded 16->32 with zeros
#pragma unroll
      for (int v = 0; v < 8; ++v) {
        a[v] = Kh[(g * 8 + v) * 128 + h * 16 + m];
        a[8 + v] = (_Float16)0.0f;
      }
      v16h b;  // B = V_tile : B[kd][n] = Vh[kd][h*16+n]
#pragma unroll
      for (int i = 0; i < 8; ++i) {
        b[i] = Vh[(g * 8 + i) * 128 + h * 16 + n];
        b[8 + i] = (_Float16)0.0f;
      }
      if (hh == 0) acc0 = WMMA_F16(a, b, acc0);
      else         acc1 = WMMA_F16(a, b, acc1);
    }
  }
  const int mo = (lane >= 16) ? 8 : 0;
#pragma unroll
  for (int j = 0; j < 8; ++j) {
    kvp[(size_t)blockIdx.x * 2048 + (w * 2 + 0) * 256 + (j + mo) * 16 + n] = acc0[j];
    kvp[(size_t)blockIdx.x * 2048 + (w * 2 + 1) * 256 + (j + mo) * 16 + n] = acc1[j];
  }
  ksp[(size_t)blockIdx.x * 128 + t] = ks;
}

// ---------- R2: final deterministic reduction ----------
__global__ void r2_reduce(const float* __restrict__ kvp, const float* __restrict__ ksp,
                          float* __restrict__ kvg, float* __restrict__ ksg) {
  int t = blockIdx.x * blockDim.x + threadIdx.x;
  if (t < 2048) {
    float s = 0.0f;
    for (int b = 0; b < 64; ++b) s += kvp[(size_t)b * 2048 + t];
    kvg[t] = s;
  } else if (t < 2176) {
    int c = t - 2048;
    float s = 0.0f;
    for (int b = 0; b < 64; ++b) s += ksp[(size_t)b * 128 + c];
    ksg[c] = s;
  }
}

// ---------- K3: attention apply + 27-tap conv + proj + LN ----------
__global__ __launch_bounds__(128) void k3_main(
    const _Float16* __restrict__ Q16, const _Float16* __restrict__ V16,
    const float* __restrict__ kvg, const float* __restrict__ ksg,
    const int* __restrict__ nbr, const _Float16* __restrict__ WdwcT,
    const float* __restrict__ b_dwc, const _Float16* __restrict__ WprojT,
    const float* __restrict__ b_proj, const float* __restrict__ sp,
    const float* __restrict__ g_n1, const float* __restrict__ b_n1,
    float* __restrict__ out) {
  __shared__ __align__(16) _Float16 Qh[16 * 128];
  __shared__ __align__(16) _Float16 kvB[2048];  // kvB[h][b][a] = kv[h][a][b]
  __shared__ __align__(16) _Float16 Ah[16 * 128];
  __shared__ float zf[128];
  __shared__ float Xf[16 * 128];
  __shared__ float Yf[16 * 128];
  __shared__ float mu[16], rsd[16];
  const int row0 = blockIdx.x * 16;
  const int t = threadIdx.x, lane = t & 31, w = t >> 5;
  const int mo = (lane >= 16) ? 8 : 0;
  {
    int r = t >> 3, c0 = (t & 7) * 16;
    const _Float16* src = Q16 + (size_t)(row0 + r) * 128 + c0;
    *(v8h*)&Qh[r * 128 + c0]     = *(const v8h*)src;
    *(v8h*)&Qh[r * 128 + c0 + 8] = *(const v8h*)(src + 8);
  }
#pragma unroll
  for (int i = 0; i < 16; ++i) {
    int e = t * 16 + i;
    int h = e >> 8, rem = e & 255, b = rem >> 4, a = rem & 15;
    kvB[e] = (_Float16)kvg[h * 256 + a * 16 + b];
  }
  __syncthreads();
  {  // z = 1/(q . ksum + eps)
    int r = t >> 3, h = t & 7;
    float s = 0.0f;
#pragma unroll
    for (int a = 0; a < 16; ++a) s += (float)Qh[r * 128 + h * 16 + a] * ksg[h * 16 + a];
    zf[r * 8 + h] = 1.0f / (s + EPS_A);
  }
  __syncthreads();
  {  // x = (q @ kv) * z   (per-head WMMA, K padded 16->32)
    const int m = lane & 15, g = (lane >> 4) & 1, n = lane & 15;
#pragma unroll
    for (int hh = 0; hh < 2; ++hh) {
      const int h = w * 2 + hh;
      v16h a;
      v8h lo = *(const v8h*)&Qh[m * 128 + h * 16 + g * 8];
#pragma unroll
      for (int v = 0; v < 8; ++v) { a[v] = lo[v]; a[8 + v] = (_Float16)0.0f; }
      v16h b;
      {
        v8h b0 = *(const v8h*)&kvB[h * 256 + n * 16 + g * 8];
#pragma unroll
        for (int i = 0; i < 8; ++i) { b[i] = b0[i]; b[8 + i] = (_Float16)0.0f; }
      }
      v8f c = zero8();
      c = WMMA_F16(a, b, c);
#pragma unroll
      for (int j = 0; j < 8; ++j) {
        int m2 = j + mo;
        Xf[m2 * 128 + h * 16 + n] = c[j] * zf[m2 * 8 + h];
      }
    }
  }
  __syncthreads();
  // conv accumulators init = x + b_dwc
  const int n0a = (w * 2) * 16, n0b = (w * 2 + 1) * 16;
  v8f acc0, acc1;
  {
    int na = n0a + (lane & 15), nb = n0b + (lane & 15);
    float bda = b_dwc[na], bdb = b_dwc[nb];
#pragma unroll
    for (int j = 0; j < 8; ++j) {
      acc0[j] = Xf[(j + mo) * 128 + na] + bda;
      acc1[j] = Xf[(j + mo) * 128 + nb] + bdb;
    }
  }
  // 27-tap gathered depthwise-block conv
  for (int tap = 0; tap < 27; ++tap) {
    __syncthreads();
    {
      int r = t >> 3, c0 = (t & 7) * 16;
      int idx = nbr[(size_t)(row0 + r) * 27 + tap];
      if (idx >= 0) {
        const _Float16* src = V16 + (size_t)idx * 128 + c0;
        *(v8h*)&Ah[r * 128 + c0]     = *(const v8h*)src;
        *(v8h*)&Ah[r * 128 + c0 + 8] = *(const v8h*)(src + 8);
      } else {
#pragma unroll
        for (int i = 0; i < 16; ++i) Ah[r * 128 + c0 + i] = (_Float16)0.0f;
      }
      if (tap < 26) __builtin_prefetch(WdwcT + (size_t)(tap + 1) * 16384, 0, 1);
    }
    __syncthreads();
    const _Float16* Bt = WdwcT + (size_t)tap * 16384;
#pragma unroll
    for (int kk = 0; kk < 128; kk += 32) {
      v16h a = load_opA_lds(Ah, 128, kk, lane);
      v16h b0 = load_opB_g(Bt, 128, n0a, kk, lane);
      acc0 = WMMA_F16(a, b0, acc0);
      v16h b1 = load_opB_g(Bt, 128, n0b, kk, lane);
      acc1 = WMMA_F16(a, b1, acc1);
    }
  }
  __syncthreads();
  {  // stage xc = x + conv as f16 A operand for proj
    int na = n0a + (lane & 15), nb = n0b + (lane & 15);
#pragma unroll
    for (int j = 0; j < 8; ++j) {
      Ah[(j + mo) * 128 + na] = (_Float16)acc0[j];
      Ah[(j + mo) * 128 + nb] = (_Float16)acc1[j];
    }
  }
  __syncthreads();
#pragma unroll
  for (int i = 0; i < 2; ++i) {  // proj + residual
    const int n0 = (i == 0) ? n0a : n0b;
    v8f a2 = zero8();
#pragma unroll
    for (int kk = 0; kk < 128; kk += 32) {
      v16h a = load_opA_lds(Ah, 128, kk, lane);
      v16h b = load_opB_g(WprojT, 128, n0, kk, lane);
      a2 = WMMA_F16(a, b, a2);
    }
    const int n = n0 + (lane & 15);
    const float bp = b_proj[n];
#pragma unroll
    for (int j = 0; j < 8; ++j) {
      int m2 = j + mo;
      Yf[m2 * 128 + n] = a2[j] + bp + sp[(size_t)(row0 + m2) * 128 + n];
    }
  }
  __syncthreads();
  if (t < 16) {  // LayerNorm stats
    float s = 0.0f;
    for (int c = 0; c < 128; ++c) s += Yf[t * 128 + c];
    float mean = s * (1.0f / 128.0f);
    float v = 0.0f;
    for (int c = 0; c < 128; ++c) { float d = Yf[t * 128 + c] - mean; v += d * d; }
    mu[t] = mean;
    rsd[t] = rsqrtf(v * (1.0f / 128.0f) + LN_EPS);
  }
  __syncthreads();
  {
    const int c = t;
    const float gg = g_n1[c], bb = b_n1[c];
#pragma unroll
    for (int r = 0; r < 16; ++r)
      out[(size_t)(row0 + r) * 128 + c] = (Yf[r * 128 + c] - mu[r]) * rsd[r] * gg + bb;
  }
}

extern "C" void kernel_launch(void* const* d_in, const int* in_sizes, int n_in,
                              void* d_out, int out_size, void* d_ws, size_t ws_size,
                              hipStream_t stream) {
  (void)in_sizes; (void)n_in; (void)out_size; (void)ws_size;
  const float* features = (const float*)d_in[0];
  const int*   indices  = (const int*)d_in[1];
  const int*   nbr      = (const int*)d_in[2];
  const float* W_pre  = (const float*)d_in[3];
  const float* b_pre  = (const float*)d_in[4];
  const float* g_pre  = (const float*)d_in[5];
  const float* be_pre = (const float*)d_in[6];
  const float* m_pre  = (const float*)d_in[7];
  const float* v_pre  = (const float*)d_in[8];
  const float* Wp1    = (const float*)d_in[9];
  const float* bp1    = (const float*)d_in[10];
  const float* g_pe   = (const float*)d_in[11];
  const float* be_pe  = (const float*)d_in[12];
  const float* m_pe   = (const float*)d_in[13];
  const float* v_pe   = (const float*)d_in[14];
  const float* Wp2    = (const float*)d_in[15];
  const float* bp2    = (const float*)d_in[16];
  const float* W_qkv  = (const float*)d_in[17];
  const float* b_qkv  = (const float*)d_in[18];
  const float* scale_p= (const float*)d_in[19];
  const float* W_dwc  = (const float*)d_in[20];
  const float* b_dwc  = (const float*)d_in[21];
  const float* W_proj = (const float*)d_in[22];
  const float* b_proj = (const float*)d_in[23];
  const float* g_n1   = (const float*)d_in[24];
  const float* b_n1   = (const float*)d_in[25];
  float* out = (float*)d_out;

  // workspace carve-up (256B aligned slabs)
  char* base = (char*)d_ws;
  size_t off = 0;
  auto alloc = [&](size_t bytes) -> char* {
    char* p = base + off;
    off = (off + bytes + 255) & ~(size_t)255;
    return p;
  };
  float*     sp_f32 = (float*)alloc((size_t)NPTS * 128 * 4);
  _Float16*  Q16    = (_Float16*)alloc((size_t)NPTS * 128 * 2);
  _Float16*  K16    = (_Float16*)alloc((size_t)NPTS * 128 * 2);
  _Float16*  V16    = (_Float16*)alloc((size_t)NPTS * 128 * 2);
  _Float16*  WpreT  = (_Float16*)alloc((size_t)128 * 64 * 2);
  _Float16*  WqkvT  = (_Float16*)alloc((size_t)384 * 128 * 2);
  _Float16*  Wp2T   = (_Float16*)alloc((size_t)128 * 128 * 2);
  _Float16*  WdwcT  = (_Float16*)alloc((size_t)27 * 128 * 128 * 2);
  _Float16*  WprojT = (_Float16*)alloc((size_t)128 * 128 * 2);
  float*     kvp    = (float*)alloc((size_t)64 * 2048 * 4);
  float*     ksp    = (float*)alloc((size_t)64 * 128 * 4);
  float*     kvg    = (float*)alloc((size_t)2048 * 4);
  float*     ksg    = (float*)alloc((size_t)128 * 4);

  // 1) weight conversion (transposed f16)
  conv_wT<<<(64 * 128 + 255) / 256, 256, 0, stream>>>(W_pre, WpreT, 64, 128, 64 * 128);
  conv_wT<<<(128 * 384 + 255) / 256, 256, 0, stream>>>(W_qkv, WqkvT, 128, 384, 128 * 384);
  conv_wT<<<(128 * 128 + 255) / 256, 256, 0, stream>>>(Wp2, Wp2T, 128, 128, 128 * 128);
  conv_wT<<<(27 * 128 * 128 + 255) / 256, 256, 0, stream>>>(W_dwc, WdwcT, 128, 128, 27 * 128 * 128);
  conv_wT<<<(128 * 128 + 255) / 256, 256, 0, stream>>>(W_proj, WprojT, 128, 128, 128 * 128);

  // 2) pipeline
  k1_pre<<<NT, 128, 0, stream>>>(features, WpreT, b_pre, g_pre, be_pre, m_pre, v_pre, sp_f32);
  k2_qkv<<<NT, 128, 0, stream>>>(sp_f32, indices, Wp1, bp1, g_pe, be_pe, m_pe, v_pe,
                                 Wp2T, bp2, WqkvT, b_qkv, scale_p, Q16, K16, V16);
  r1_kv<<<64, 128, 0, stream>>>(K16, V16, kvp, ksp);
  r2_reduce<<<9, 256, 0, stream>>>(kvp, ksp, kvg, ksg);
  k3_main<<<NT, 128, 0, stream>>>(Q16, V16, kvg, ksg, nbr, WdwcT, b_dwc, WprojT, b_proj,
                                  sp_f32, g_n1, b_n1, out);
}
